// SelfAttention_14370960573166
// MI455X (gfx1250) — matrix operand order repaired and checked by reading, not compiled, not run on hardware
//
#include <hip/hip_runtime.h>
#include <math.h>

// ---------------------------------------------------------------------------
// Self-attention forward for MI455X (gfx1250, wave32, WMMA f16 16x16x32).
// b=4, s=2048, d=1024, heads=16, d_head=64, causal.
// Attention K/V tiles staged once per WG into LDS via async global->LDS copies
// (ASYNCcnt), double-buffered against the WMMA pipeline.
// ---------------------------------------------------------------------------

typedef __attribute__((ext_vector_type(16))) _Float16 v16h;
typedef __attribute__((ext_vector_type(8)))  _Float16 v8h;
typedef __attribute__((ext_vector_type(8)))  float    v8f;
typedef __attribute__((ext_vector_type(4)))  int      v4i_t;

union V16H { v16h v; v8h h[2]; };

#define B_SZ   4
#define S_SZ   2048
#define D_SZ   1024
#define H_SZ   16
#define DH_SZ  64
#define M_SZ   (B_SZ * S_SZ)        // 8192 rows
#define LOG2E  1.44269504088896341f

// D = A(16x32 f16) * B(32x16 f16) + C(16x16 f32)
__device__ __forceinline__ v8f wmma_f16(v16h a, v16h b, v8f c) {
    return __builtin_amdgcn_wmma_f32_16x16x32_f16(
        /*neg_a=*/false, a, /*neg_b=*/false, b,
        /*c_mod=*/(short)0, c, /*reuse_a=*/false, /*reuse_b=*/false);
}

// A-fragment: 16(M) x 32(K). Lane: M = lane&15. Element e -> K = k0
// + 16*(lane>>4) + e. Two contiguous 16B chunks per lane.
__device__ __forceinline__ v16h load_afrag(const _Float16* __restrict__ base,
                                           int row0, int ld, int k0, int lane) {
    const int m = lane & 15, half = lane >> 4;
    const _Float16* p = base + (size_t)(row0 + m) * ld + k0 + 16 * half;
    V16H u;
    u.h[0] = *(const v8h*)(p);
    u.h[1] = *(const v8h*)(p + 8);
    return u.v;
}

// B-fragment: 32(K) x 16(N), where memory holds rows indexed by N with K
// contiguous (i.e. we read row n of W for W^T). Lane: N = col0 + (lane&15),
// element e -> K = k0 + 16*(lane>>4) + e.  One contiguous 32B load per lane.
__device__ __forceinline__ v16h load_bfrag(const _Float16* __restrict__ base,
                                           int col0, int ld, int k0, int lane) {
    const int n = lane & 15, half = lane >> 4;
    return *(const v16h*)(base + (size_t)(col0 + n) * ld + k0 + 16 * half);
}

// --- CDNA5 async global->LDS copy (16 bytes per lane), ASYNCcnt tracked -----
__device__ __forceinline__ void async_cp16(const _Float16* g, _Float16* l) {
#if __has_builtin(__builtin_amdgcn_global_load_async_to_lds_b128)
    __builtin_amdgcn_global_load_async_to_lds_b128(
        (__attribute__((address_space(1))) v4i_t*)(_Float16*)g,
        (__attribute__((address_space(3))) v4i_t*)l,
        0, 0);
#else
    unsigned int loff = (unsigned int)(unsigned long long)l;  // LDS byte offset
    asm volatile("global_load_async_to_lds_b128 %0, %1, off"
                 :: "v"(loff), "v"((unsigned long long)g) : "memory");
#endif
}

__device__ __forceinline__ void wait_async_le2() {
#if __has_builtin(__builtin_amdgcn_s_wait_asynccnt)
    __builtin_amdgcn_s_wait_asynccnt(2);
#else
    asm volatile("s_wait_asynccnt 0x2" ::: "memory");
#endif
}
__device__ __forceinline__ void wait_async_le0() {
#if __has_builtin(__builtin_amdgcn_s_wait_asynccnt)
    __builtin_amdgcn_s_wait_asynccnt(0);
#else
    asm volatile("s_wait_asynccnt 0x0" ::: "memory");
#endif
}

// ---------------------------------------------------------------------------
__global__ __launch_bounds__(256) void f32_to_f16_kernel(
        const float* __restrict__ in, _Float16* __restrict__ out, int n) {
    for (int i = blockIdx.x * blockDim.x + threadIdx.x; i < n;
         i += gridDim.x * blockDim.x)
        out[i] = (_Float16)in[i];
}

// ---------------------------------------------------------------------------
// QKV projection: qkv[m,n] = sum_k xh[m,k]*w1h[n,k] + b1[n], n in [0,3072).
// Writes Q,K as f16 [b,h,s,64] and V transposed [b,h,64,s].
// ---------------------------------------------------------------------------
__global__ __launch_bounds__(256) void qkv_gemm_kernel(
        const _Float16* __restrict__ xh,   // [8192,1024]
        const _Float16* __restrict__ w1h,  // [3072,1024]
        const float*    __restrict__ b1,   // [3072]
        _Float16* __restrict__ Qh, _Float16* __restrict__ Kh,
        _Float16* __restrict__ Vt) {
    const int lane = threadIdx.x & 31;
    const int w    = threadIdx.x >> 5;
    const int m0 = blockIdx.x * 64  + (w >> 2) * 32;
    const int n0 = blockIdx.y * 256 + (w & 3) * 64;

    v8f acc[2][4];
#pragma unroll
    for (int mi = 0; mi < 2; ++mi)
#pragma unroll
        for (int ni = 0; ni < 4; ++ni) acc[mi][ni] = (v8f){0,0,0,0,0,0,0,0};

    for (int k0 = 0; k0 < D_SZ; k0 += 32) {
        v16h a0 = load_afrag(xh, m0,      D_SZ, k0, lane);
        v16h a1 = load_afrag(xh, m0 + 16, D_SZ, k0, lane);
#pragma unroll
        for (int ni = 0; ni < 4; ++ni) {
            v16h b = load_bfrag(w1h, n0 + 16 * ni, D_SZ, k0, lane);
            acc[0][ni] = wmma_f16(a0, b, acc[0][ni]);
            acc[1][ni] = wmma_f16(a1, b, acc[1][ni]);
        }
    }

    const int nl = lane & 15, half = lane >> 4;
#pragma unroll
    for (int ni = 0; ni < 4; ++ni) {
        const int n    = n0 + 16 * ni + nl;
        const float bv = b1[n];
        const int sec  = n >> 10;         // 0:Q 1:K 2:V (uniform per WG)
        const int nn   = n & 1023;
        const int head = nn >> 6;
        const int dh   = nn & 63;
#pragma unroll
        for (int mi = 0; mi < 2; ++mi) {
            const int mbase = m0 + 16 * mi + 8 * half;   // 8 consecutive rows
            const int bq    = mbase >> 11;               // batch (uniform)
            const int srow  = mbase & (S_SZ - 1);
            if (sec == 2) {
                v8h pk;
#pragma unroll
                for (int r = 0; r < 8; ++r)
                    pk[r] = (_Float16)(acc[mi][ni][r] + bv);
                size_t addr = (((size_t)bq * H_SZ + head) * DH_SZ + dh) * S_SZ + srow;
                *(v8h*)(Vt + addr) = pk;
            } else {
                _Float16* dst = (sec == 0) ? Qh : Kh;
                size_t rbase = (((size_t)bq * H_SZ + head) * S_SZ + srow) * DH_SZ + dh;
#pragma unroll
                for (int r = 0; r < 8; ++r)
                    dst[rbase + (size_t)r * DH_SZ] =
                        (_Float16)(acc[mi][ni][r] + bv);
            }
        }
    }
}

// ---------------------------------------------------------------------------
// Causal flash attention, computed transposed so each lane owns one query:
//   S^T = K * Q^T          (A = K tile from LDS, B = Q^T from registers)
//   O^T = V^T * P^T        (A = V^T tile from LDS, B = P^T built by shuffle)
// WG (8 waves, 128 queries) shares K/V tiles via async global->LDS staging,
// double-buffered. Per-wave causal predicate is uniform -> EXEC all-ones.
// ---------------------------------------------------------------------------
__global__ __launch_bounds__(256) void attn_kernel(
        const _Float16* __restrict__ Qh,  // [b,h,s,64]
        const _Float16* __restrict__ Kh,  // [b,h,s,64]
        const _Float16* __restrict__ Vt,  // [b,h,64,s]
        _Float16* __restrict__ Oh) {      // [b,s,1024]
    __shared__ __align__(16) _Float16 ldsK[2][32][72];  // 32 keys x 64 d (+pad)
    __shared__ __align__(16) _Float16 ldsV[2][64][40];  // 64 d x 32 keys (+pad)

    const int tid  = threadIdx.x;
    const int lane = tid & 31;
    const int w    = tid >> 5;
    const int bh   = blockIdx.y;            // b*16 + head
    const int bq   = bh >> 4, head = bh & 15;
    const int Q0   = blockIdx.x * 128;
    const int q0   = Q0 + w * 16;
    const int nl   = lane & 15, half = lane >> 4;

    const _Float16* Q = Qh + (size_t)bh * S_SZ * DH_SZ;
    const _Float16* K = Kh + (size_t)bh * S_SZ * DH_SZ;
    const _Float16* V = Vt + (size_t)bh * DH_SZ * S_SZ;

    // Stage one 32-key block (K: 32x64, V^T: 64x32) into LDS buffer `buf`.
    // Each thread issues exactly 2 async 16B copies.
    auto stage = [&](int kb, int buf) {
        const int kr = tid >> 3, kc = (tid & 7) * 8;     // K: 32 rows x 8 chunks
        async_cp16(K + (size_t)(kb + kr) * DH_SZ + kc, &ldsK[buf][kr][kc]);
        const int vr = tid >> 2, vc = (tid & 3) * 8;     // V: 64 rows x 4 chunks
        async_cp16(V + (size_t)vr * S_SZ + kb + vc, &ldsV[buf][vr][vc]);
    };

    // Q^T B-fragments: lane holds query q0+nl, d contiguous.
    const v16h qb0 = *(const v16h*)(Q + (size_t)(q0 + nl) * DH_SZ + 16 * half);
    const v16h qb1 = *(const v16h*)(Q + (size_t)(q0 + nl) * DH_SZ + 32 + 16 * half);

    v8f oacc[4];
#pragma unroll
    for (int t = 0; t < 4; ++t) oacc[t] = (v8f){0,0,0,0,0,0,0,0};
    float m_run = -INFINITY, l_run = 0.0f;

    const int qg   = q0 + nl;               // this lane's query index
    const int kend = q0 + 16;               // causal: this wave needs keys < kend
    const int kmax = Q0 + 128;              // WG-wide key range (uniform)

    stage(0, 0);                            // prologue: block 0 -> buf 0
    int buf = 0;
    for (int kb = 0; kb < kmax; kb += 32, buf ^= 1) {
        const bool more = (kb + 32) < kmax;
        if (more) stage(kb + 32, buf ^ 1);  // overlap next block's copy
        if (more) wait_async_le2();         // oldest 2 (current buf) done
        else      wait_async_le0();
        __syncthreads();                    // LDS tiles visible to all waves

        if (kb < kend) {                    // wave-uniform causal predicate
            // S^T tiles from LDS: tile0 keys kb..kb+15, tile1 kb+16..kb+31
            V16H ka0a, ka0b, ka1a, ka1b;
            ka0a.h[0] = *(const v8h*)&ldsK[buf][nl][16 * half];
            ka0a.h[1] = *(const v8h*)&ldsK[buf][nl][16 * half + 8];
            ka0b.h[0] = *(const v8h*)&ldsK[buf][nl][32 + 16 * half];
            ka0b.h[1] = *(const v8h*)&ldsK[buf][nl][32 + 16 * half + 8];
            ka1a.h[0] = *(const v8h*)&ldsK[buf][16 + nl][16 * half];
            ka1a.h[1] = *(const v8h*)&ldsK[buf][16 + nl][16 * half + 8];
            ka1b.h[0] = *(const v8h*)&ldsK[buf][16 + nl][32 + 16 * half];
            ka1b.h[1] = *(const v8h*)&ldsK[buf][16 + nl][32 + 16 * half + 8];

            v8f s0 = (v8f){0,0,0,0,0,0,0,0};
            v8f s1 = (v8f){0,0,0,0,0,0,0,0};
            s0 = wmma_f16(ka0a.v, qb0, s0);  s0 = wmma_f16(ka0b.v, qb1, s0);
            s1 = wmma_f16(ka1a.v, qb0, s1);  s1 = wmma_f16(ka1b.v, qb1, s1);

            // scale 1/sqrt(64), causal mask, per-query (per-lane) block max.
            float p0[8], p1[8];
            float mblk = -INFINITY;
#pragma unroll
            for (int r = 0; r < 8; ++r) {
                const int kg0 = kb + 8 * half + r;
                p0[r] = (kg0      <= qg) ? s0[r] * 0.125f : -INFINITY;
                p1[r] = (kg0 + 16 <= qg) ? s1[r] * 0.125f : -INFINITY;
                mblk = fmaxf(mblk, fmaxf(p0[r], p1[r]));
            }
            mblk = fmaxf(mblk, __shfl_xor(mblk, 16, 32));
            const float m_new = fmaxf(m_run, mblk);
            const float scale = exp2f((m_run - m_new) * LOG2E);

            float lsum = 0.0f;
#pragma unroll
            for (int r = 0; r < 8; ++r) {
                p0[r] = exp2f((p0[r] - m_new) * LOG2E);
                p1[r] = exp2f((p1[r] - m_new) * LOG2E);
                lsum += p0[r] + p1[r];
            }
            lsum += __shfl_xor(lsum, 16, 32);
            l_run = l_run * scale + lsum;
            m_run = m_new;
#pragma unroll
            for (int t = 0; t < 4; ++t) oacc[t] *= scale;

            // P^T B-fragment (32 keys x 16 queries): 8 local + 8 via xor-16.
            V16H pb;
#pragma unroll
            for (int r = 0; r < 8; ++r) {
                const float send = half ? p0[r] : p1[r];
                const float recv = __shfl_xor(send, 16, 32);
                pb.h[0][r] = (_Float16)(half ? recv  : p0[r]);
                pb.h[1][r] = (_Float16)(half ? p1[r] : recv);
            }

            // O^T += V^T * P^T  (4 d-tiles of 16) from LDS
#pragma unroll
            for (int t = 0; t < 4; ++t) {
                V16H va;
                va.h[0] = *(const v8h*)&ldsV[buf][16 * t + nl][16 * half];
                va.h[1] = *(const v8h*)&ldsV[buf][16 * t + nl][16 * half + 8];
                oacc[t] = wmma_f16(va.v, pb.v, oacc[t]);
            }
        }
        __syncthreads();                    // buf may be overwritten next iter
    }

    const float inv_l = 1.0f / l_run;
    // O^T lane holds query q0+nl, d = 16*t + 8*half + r (8 contiguous per t).
    const size_t obase =
        ((size_t)bq * S_SZ + q0 + nl) * D_SZ + head * DH_SZ;
#pragma unroll
    for (int t = 0; t < 4; ++t) {
        v8h pk;
#pragma unroll
        for (int r = 0; r < 8; ++r) pk[r] = (_Float16)(oacc[t][r] * inv_l);
        *(v8h*)(Oh + obase + 16 * t + 8 * half) = pk;
    }
}

// ---------------------------------------------------------------------------
// Output projection: out[m,n] = sum_k Oh[m,k]*w2h[n,k] + b2[n]  (f32 out)
// ---------------------------------------------------------------------------
__global__ __launch_bounds__(256) void out_gemm_kernel(
        const _Float16* __restrict__ Ah,   // [8192,1024] f16
        const _Float16* __restrict__ w2h,  // [1024,1024] f16
        const float*    __restrict__ b2,   // [1024]
        float* __restrict__ out) {         // [8192,1024] f32
    const int lane = threadIdx.x & 31;
    const int w    = threadIdx.x >> 5;
    const int m0 = blockIdx.x * 64  + (w >> 2) * 32;
    const int n0 = blockIdx.y * 256 + (w & 3) * 64;

    v8f acc[2][4];
#pragma unroll
    for (int mi = 0; mi < 2; ++mi)
#pragma unroll
        for (int ni = 0; ni < 4; ++ni) acc[mi][ni] = (v8f){0,0,0,0,0,0,0,0};

    for (int k0 = 0; k0 < D_SZ; k0 += 32) {
        v16h a0 = load_afrag(Ah, m0,      D_SZ, k0, lane);
        v16h a1 = load_afrag(Ah, m0 + 16, D_SZ, k0, lane);
#pragma unroll
        for (int ni = 0; ni < 4; ++ni) {
            v16h b = load_bfrag(w2h, n0 + 16 * ni, D_SZ, k0, lane);
            acc[0][ni] = wmma_f16(a0, b, acc[0][ni]);
            acc[1][ni] = wmma_f16(a1, b, acc[1][ni]);
        }
    }

    const int nl = lane & 15, half = lane >> 4;
#pragma unroll
    for (int ni = 0; ni < 4; ++ni) {
        const int n    = n0 + 16 * ni + nl;
        const float bv = b2[n];
#pragma unroll
        for (int mi = 0; mi < 2; ++mi) {
            const int mbase = m0 + 16 * mi + 8 * half;
#pragma unroll
            for (int r = 0; r < 8; ++r)
                out[(size_t)(mbase + r) * D_SZ + n] = acc[mi][ni][r] + bv;
        }
    }
}

// ---------------------------------------------------------------------------
extern "C" void kernel_launch(void* const* d_in, const int* in_sizes, int n_in,
                              void* d_out, int out_size, void* d_ws, size_t ws_size,
                              hipStream_t stream) {
    const float* x   = (const float*)d_in[0];  // [4,2048,1024]
    const float* w1  = (const float*)d_in[1];  // [3072,1024]
    const float* b1  = (const float*)d_in[2];  // [3072]
    const float* w2  = (const float*)d_in[3];  // [1024,1024]
    const float* b2  = (const float*)d_in[4];  // [1024]
    float* out = (float*)d_out;

    char* ws = (char*)d_ws;
    const size_t n_x  = (size_t)M_SZ * D_SZ;        // 8388608
    const size_t n_w1 = (size_t)3 * D_SZ * D_SZ;    // 3145728
    const size_t n_w2 = (size_t)D_SZ * D_SZ;        // 1048576
    _Float16* xh  = (_Float16*)ws;                      ws += n_x  * 2;
    _Float16* w1h = (_Float16*)ws;                      ws += n_w1 * 2;
    _Float16* w2h = (_Float16*)ws;                      ws += n_w2 * 2;
    _Float16* Qh  = (_Float16*)ws;                      ws += n_x  * 2;
    _Float16* Kh  = (_Float16*)ws;                      ws += n_x  * 2;
    _Float16* Vt  = (_Float16*)ws;                      ws += n_x  * 2;
    _Float16* Oh  = (_Float16*)ws;

    f32_to_f16_kernel<<<2048, 256, 0, stream>>>(x,  xh,  (int)n_x);
    f32_to_f16_kernel<<<2048, 256, 0, stream>>>(w1, w1h, (int)n_w1);
    f32_to_f16_kernel<<<1024, 256, 0, stream>>>(w2, w2h, (int)n_w2);

    qkv_gemm_kernel<<<dim3(M_SZ / 64, 3072 / 256), 256, 0, stream>>>(
        xh, w1h, b1, Qh, Kh, Vt);

    attn_kernel<<<dim3(S_SZ / 128, B_SZ * H_SZ), 256, 0, stream>>>(
        Qh, Kh, Vt, Oh);

    out_gemm_kernel<<<dim3(M_SZ / 64, D_SZ / 256), 256, 0, stream>>>(
        Oh, w2h, b2, out);
}
